// MolEncoder_40303973106171
// MI455X (gfx1250) — matrix operand order repaired and checked by reading, not compiled, not run on hardware
//
#include <hip/hip_runtime.h>
#include <hip/hip_bf16.h>
#include <math.h>

#define N_NODES 50000
#define N_EDGES 800000
#define ETOT    (N_EDGES + N_NODES)   // self loops appended
#define F_IN    64
#define HIDDEN  128
#define HEADS   4
#define CHN     32                    // HIDDEN / HEADS
#define OUTF    256
#define N_GRAPHS 1024
#define BN_EPS  1e-5f
#define SLOPE   0.2f

typedef __bf16 bf16;
typedef bf16  v16bf __attribute__((ext_vector_type(16)));
typedef float v8f   __attribute__((ext_vector_type(8)));
typedef int   v4i   __attribute__((vector_size(16)));   // async-copy operand type

#if defined(__has_builtin)
#if __has_builtin(__builtin_amdgcn_global_load_async_to_lds_b128)
#define HAVE_ASYNC_LDS 1
#endif
#endif

static inline int cdiv(int a, int b) { return (a + b - 1) / b; }

// ---------- helpers ----------
__device__ inline unsigned short f2bf(float f) {
  unsigned u = __float_as_uint(f);
  unsigned r = u + 0x7fffu + ((u >> 16) & 1u);   // round-to-nearest-even
  return (unsigned short)(r >> 16);
}
__device__ inline unsigned ordkey(float f) {     // monotone float->uint
  unsigned u = __float_as_uint(f);
  return (u & 0x80000000u) ? ~u : (u | 0x80000000u);
}
__device__ inline float ordval(unsigned k) {
  unsigned u = (k & 0x80000000u) ? (k & 0x7fffffffu) : ~k;
  return __uint_as_float(u);
}
__device__ inline void edge_sd(const int* __restrict__ ei, int e, int& s, int& d) {
  if (e < N_EDGES) { s = ei[e]; d = ei[N_EDGES + e]; }
  else             { s = d = e - N_EDGES; }
}

// ---------- f32 -> bf16 conversion ----------
__global__ void k_cvt_bf16(const float* __restrict__ src, unsigned short* __restrict__ dst, int n) {
  int i = blockIdx.x * blockDim.x + threadIdx.x;
  if (i < n) dst[i] = f2bf(src[i]);
}

// ---------- WMMA GEMM: D[M,N] = A[M,K](bf16) * B[K,N](bf16) (+bias[N]) ----------
// Block = 256 threads (8 waves). Block tile = 16 rows x 128 cols; each wave
// computes one 16x16 tile. A(16xK) staged row-major (async-to-LDS when
// available); B(Kx128) staged *swizzled* so each lane's B fragment is a
// contiguous 32B run -> inner loop is 4x ds_load_b128 + 1x v_wmma per k-step,
// K fully unrolled.
template <int K>
__global__ __launch_bounds__(256) void k_gemm_bf16(const unsigned short* __restrict__ A,
                                                   const unsigned short* __restrict__ B,
                                                   float* __restrict__ D,
                                                   const float* __restrict__ bias,
                                                   int N) {
  constexpr int NPANEL = 128;
  // Bsw layout: element (k, c) at ((k>>4)*NPANEL + c)*16 + (k&15)
  __shared__ unsigned short Bsw[K * NPANEL];  // <= 32 KB
  __shared__ unsigned short As[16 * K];       // <= 4 KB

  const int tid = threadIdx.x;
  const int colbase = blockIdx.x * NPANEL;    // N col panel
  const int rowbase = blockIdx.y * 16;        // M row tile

  // ---- stage A tile [16 x K], row-major ----
  for (int idx = tid * 8; idx < 16 * K; idx += 256 * 8) {
    int r = idx / K;
    int c = idx % K;
#ifdef HAVE_ASYNC_LDS
    __builtin_amdgcn_global_load_async_to_lds_b128(
        (v4i*)&A[(size_t)(rowbase + r) * K + c], (v4i*)&As[idx], 0, 0);
#else
    *(uint4*)&As[idx] = *(const uint4*)&A[(size_t)(rowbase + r) * K + c];
#endif
  }
  // ---- stage B panel [K x 128], swizzled to fragment order ----
  for (int idx = tid * 8; idx < K * NPANEL; idx += 256 * 8) {
    int k = idx >> 7;          // / NPANEL
    int c = idx & (NPANEL - 1);
    uint4 q = *(const uint4*)&B[(size_t)k * N + colbase + c];
    const unsigned short* hs = (const unsigned short*)&q;
    int base = ((k >> 4) * NPANEL) * 16 + (k & 15);
#pragma unroll
    for (int t = 0; t < 8; ++t)
      Bsw[base + (c + t) * 16] = hs[t];
  }
#ifdef HAVE_ASYNC_LDS
  asm volatile("s_wait_asynccnt 0" ::: "memory");
#endif
  __syncthreads();

  const int wv   = tid >> 5;          // 0..7 : which 16-col tile in the panel
  const int lane = tid & 31;
  const int l15  = lane & 15;
  const int khalfA = (lane & 16) ? 16 : 0;  // A frag: lanes 16-31 hold K+16
  const int khalfB = (lane & 16) ? 16 : 0;  // B frag: lanes 16-31 hold K+16
  const int cpan = wv * 16 + l15;           // column within panel

  v8f acc = {};
#pragma unroll
  for (int kb = 0; kb < K; kb += 32) {
    // A fragment: one contiguous 16-half run of row l15, K = kb+khalfA..+15
    union { uint4 q[2]; v16bf v; } ua;
    ua.q[0] = *(const uint4*)&As[l15 * K + kb + khalfA];
    ua.q[1] = *(const uint4*)&As[l15 * K + kb + 8 + khalfA];
    // B fragment: one contiguous 16-half run of column cpan, K = kb+khalfB..+15
    union { uint4 q[2]; v16bf v; } ub;
    int bidx = (((kb + khalfB) >> 4) * NPANEL + cpan) * 16;
    ub.q[0] = *(const uint4*)&Bsw[bidx];
    ub.q[1] = *(const uint4*)&Bsw[bidx + 8];
    acc = __builtin_amdgcn_wmma_f32_16x16x32_bf16(false, ua.v, false, ub.v, (short)0, acc,
                                                  false, false);
  }

  const int ncol = colbase + cpan;
  float bv = 0.f;
  if (bias) bv = bias[ncol];
#pragma unroll
  for (int r = 0; r < 8; ++r) {
    int row = rowbase + r + ((lane & 16) ? 8 : 0);
    D[(size_t)row * N + ncol] = acc[r] + bv;
  }
}

// ---------- per-(node,head) attention scores ----------
__global__ void k_edge_scores(const float* __restrict__ h,
                              const float* __restrict__ asrc,
                              const float* __restrict__ adst,
                              float* __restrict__ es, float* __restrict__ ed) {
  int i = blockIdx.x * blockDim.x + threadIdx.x;
  if (i >= N_NODES * HEADS) return;
  int n = i >> 2, hh = i & 3;
  const float* hp = h + (size_t)n * HIDDEN + hh * CHN;
  const float* as = asrc + hh * CHN;
  const float* ad = adst + hh * CHN;
  float s1 = 0.f, s2 = 0.f;
#pragma unroll 8
  for (int c = 0; c < CHN; ++c) { float v = hp[c]; s1 += v * as[c]; s2 += v * ad[c]; }
  es[i] = s1; ed[i] = s2;
}

__global__ void k_init_maxdenom(unsigned* __restrict__ mkey, float* __restrict__ denom) {
  int i = blockIdx.x * blockDim.x + threadIdx.x;
  if (i < N_NODES * HEADS) { mkey[i] = 0u; denom[i] = 0.f; }
}

__global__ void k_edge_max(const int* __restrict__ ei,
                           const float* __restrict__ es, const float* __restrict__ ed,
                           unsigned* __restrict__ mkey) {
  int i = blockIdx.x * blockDim.x + threadIdx.x;
  if (i >= ETOT * HEADS) return;
  int e = i >> 2, hh = i & 3, s, d;
  edge_sd(ei, e, s, d);
  float v = es[s * HEADS + hh] + ed[d * HEADS + hh];
  float l = v > 0.f ? v : SLOPE * v;
  atomicMax(&mkey[d * HEADS + hh], ordkey(l));
}

__global__ void k_edge_exp(const int* __restrict__ ei,
                           const float* __restrict__ es, const float* __restrict__ ed,
                           const unsigned* __restrict__ mkey,
                           float* __restrict__ denom, float* __restrict__ exbuf) {
  int i = blockIdx.x * blockDim.x + threadIdx.x;
  if (i >= ETOT * HEADS) return;
  int e = i >> 2, hh = i & 3, s, d;
  edge_sd(ei, e, s, d);
  float v = es[s * HEADS + hh] + ed[d * HEADS + hh];
  float l = v > 0.f ? v : SLOPE * v;
  float ex = expf(l - ordval(mkey[d * HEADS + hh]));
  exbuf[i] = ex;
  atomicAdd(&denom[d * HEADS + hh], ex);
}

__global__ void k_init_bias(float* __restrict__ attout, const float* __restrict__ bias) {
  int i = blockIdx.x * blockDim.x + threadIdx.x;
  if (i < N_NODES * HIDDEN) attout[i] = bias[i & (HIDDEN - 1)];
}

// one block (128 thr) per edge: thread = (head, channel)
__global__ void k_edge_agg(const int* __restrict__ ei,
                           const float* __restrict__ exbuf,
                           const float* __restrict__ denom,
                           const float* __restrict__ h,
                           float* __restrict__ attout) {
  int e = blockIdx.x, s, d;
  edge_sd(ei, e, s, d);
  int hh = threadIdx.x >> 5, c = threadIdx.x & 31;
  float alpha = exbuf[e * HEADS + hh] / (denom[d * HEADS + hh] + 1e-16f);
  float v = alpha * h[(size_t)s * HIDDEN + hh * CHN + c];
  atomicAdd(&attout[(size_t)d * HIDDEN + hh * CHN + c], v);
}

// ---------- batchnorm stats: one block per channel ----------
__global__ void k_bn_stats(const float* __restrict__ x, float* __restrict__ mu, float* __restrict__ var) {
  int c = blockIdx.x;
  float s = 0.f, s2 = 0.f;
  for (int n = threadIdx.x; n < N_NODES; n += blockDim.x) {
    float v = x[(size_t)n * HIDDEN + c];
    s += v; s2 += v * v;
  }
  __shared__ float sh[256], sh2[256];
  sh[threadIdx.x] = s; sh2[threadIdx.x] = s2;
  __syncthreads();
  for (int off = blockDim.x >> 1; off > 0; off >>= 1) {
    if (threadIdx.x < off) { sh[threadIdx.x] += sh[threadIdx.x + off]; sh2[threadIdx.x] += sh2[threadIdx.x + off]; }
    __syncthreads();
  }
  if (threadIdx.x == 0) {
    float m = sh[0] / (float)N_NODES;
    mu[c] = m;
    var[c] = sh2[0] / (float)N_NODES - m * m;
  }
}

__global__ void k_bn_elu(const float* __restrict__ x,
                         const float* __restrict__ mu, const float* __restrict__ var,
                         const float* __restrict__ g, const float* __restrict__ b,
                         float* __restrict__ yf, unsigned short* __restrict__ y16) {
  int i = blockIdx.x * blockDim.x + threadIdx.x;
  if (i >= N_NODES * HIDDEN) return;
  int c = i & (HIDDEN - 1);
  float v = (x[i] - mu[c]) * rsqrtf(var[c] + BN_EPS) * g[c] + b[c];
  float y = v > 0.f ? v : expm1f(v);
  yf[i] = y;
  y16[i] = f2bf(y);
}

// ---------- pooling ----------
__global__ void k_pool_zero(float* __restrict__ pooled, float* __restrict__ cnt) {
  int i = blockIdx.x * blockDim.x + threadIdx.x;
  if (i < N_GRAPHS * HIDDEN) pooled[i] = 0.f;
  if (i < N_GRAPHS) cnt[i] = 0.f;
}
__global__ void k_pool_acc(const float* __restrict__ feat, const int* __restrict__ batch,
                           float* __restrict__ pooled, float* __restrict__ cnt) {
  int i = blockIdx.x * blockDim.x + threadIdx.x;
  if (i >= N_NODES * HIDDEN) return;
  int n = i >> 7, c = i & (HIDDEN - 1);
  int gidx = batch[n];
  atomicAdd(&pooled[(size_t)gidx * HIDDEN + c], feat[i]);
  if (c == 0) atomicAdd(&cnt[gidx], 1.0f);
}
__global__ void k_pool_div(float* __restrict__ pooled, const float* __restrict__ cnt,
                           unsigned short* __restrict__ p16) {
  int i = blockIdx.x * blockDim.x + threadIdx.x;
  if (i >= N_GRAPHS * HIDDEN) return;
  float v = pooled[i] / fmaxf(cnt[i >> 7], 1.0f);
  pooled[i] = v;
  p16[i] = f2bf(v);
}

// ---------- host side ----------
struct Ws {
  unsigned short *x16, *h16, *W1_16, *W2_16, *Wl16, *p16;
  float *bufA, *bufB, *es, *ed, *denom, *exbuf, *mu, *var, *pooled, *cnt;
  unsigned *mkey;
};

static char* carve(char*& p, size_t bytes) {
  char* r = p;
  p += (bytes + 255) & ~(size_t)255;
  return r;
}

static void run_edge_bn(const float* gemm_out, const float* asrc, const float* adst,
                        const float* bias, const float* bng, const float* bnb,
                        float* att_out, float* out_f32, unsigned short* out16,
                        const int* ei, Ws& w, hipStream_t stream) {
  k_edge_scores<<<cdiv(N_NODES * HEADS, 256), 256, 0, stream>>>(gemm_out, asrc, adst, w.es, w.ed);
  k_init_maxdenom<<<cdiv(N_NODES * HEADS, 256), 256, 0, stream>>>(w.mkey, w.denom);
  k_edge_max<<<cdiv(ETOT * HEADS, 256), 256, 0, stream>>>(ei, w.es, w.ed, w.mkey);
  k_edge_exp<<<cdiv(ETOT * HEADS, 256), 256, 0, stream>>>(ei, w.es, w.ed, w.mkey, w.denom, w.exbuf);
  k_init_bias<<<cdiv(N_NODES * HIDDEN, 256), 256, 0, stream>>>(att_out, bias);
  k_edge_agg<<<ETOT, HEADS * CHN, 0, stream>>>(ei, w.exbuf, w.denom, gemm_out, att_out);
  k_bn_stats<<<HIDDEN, 256, 0, stream>>>(att_out, w.mu, w.var);
  k_bn_elu<<<cdiv(N_NODES * HIDDEN, 256), 256, 0, stream>>>(att_out, w.mu, w.var, bng, bnb,
                                                            out_f32, out16);
}

extern "C" void kernel_launch(void* const* d_in, const int* in_sizes, int n_in,
                              void* d_out, int out_size, void* d_ws, size_t ws_size,
                              hipStream_t stream) {
  const float* x     = (const float*)d_in[0];
  const int*   ei    = (const int*)d_in[1];
  const int*   batch = (const int*)d_in[2];
  const float* W1    = (const float*)d_in[3];
  const float* as1   = (const float*)d_in[4];
  const float* ad1   = (const float*)d_in[5];
  const float* b1    = (const float*)d_in[6];
  const float* W2    = (const float*)d_in[7];
  const float* as2   = (const float*)d_in[8];
  const float* ad2   = (const float*)d_in[9];
  const float* b2    = (const float*)d_in[10];
  const float* bn1g  = (const float*)d_in[11];
  const float* bn1b  = (const float*)d_in[12];
  const float* bn2g  = (const float*)d_in[13];
  const float* bn2b  = (const float*)d_in[14];
  const float* Wlin  = (const float*)d_in[15];
  const float* blin  = (const float*)d_in[16];
  float* out = (float*)d_out;

  char* p = (char*)d_ws;
  Ws w;
  w.x16    = (unsigned short*)carve(p, (size_t)N_NODES * F_IN * 2);
  w.h16    = (unsigned short*)carve(p, (size_t)N_NODES * HIDDEN * 2);
  w.W1_16  = (unsigned short*)carve(p, (size_t)F_IN * HIDDEN * 2);
  w.W2_16  = (unsigned short*)carve(p, (size_t)HIDDEN * HIDDEN * 2);
  w.Wl16   = (unsigned short*)carve(p, (size_t)HIDDEN * OUTF * 2);
  w.p16    = (unsigned short*)carve(p, (size_t)N_GRAPHS * HIDDEN * 2);
  w.bufA   = (float*)carve(p, (size_t)N_NODES * HIDDEN * 4);
  w.bufB   = (float*)carve(p, (size_t)N_NODES * HIDDEN * 4);
  w.es     = (float*)carve(p, (size_t)N_NODES * HEADS * 4);
  w.ed     = (float*)carve(p, (size_t)N_NODES * HEADS * 4);
  w.denom  = (float*)carve(p, (size_t)N_NODES * HEADS * 4);
  w.mkey   = (unsigned*)carve(p, (size_t)N_NODES * HEADS * 4);
  w.exbuf  = (float*)carve(p, (size_t)ETOT * HEADS * 4);
  w.mu     = (float*)carve(p, HIDDEN * 4);
  w.var    = (float*)carve(p, HIDDEN * 4);
  w.pooled = (float*)carve(p, (size_t)N_GRAPHS * HIDDEN * 4);
  w.cnt    = (float*)carve(p, (size_t)N_GRAPHS * 4);

  // bf16 conversions of x and all weight matrices
  k_cvt_bf16<<<cdiv(N_NODES * F_IN, 256), 256, 0, stream>>>(x, w.x16, N_NODES * F_IN);
  k_cvt_bf16<<<cdiv(F_IN * HIDDEN, 256), 256, 0, stream>>>(W1, w.W1_16, F_IN * HIDDEN);
  k_cvt_bf16<<<cdiv(HIDDEN * HIDDEN, 256), 256, 0, stream>>>(W2, w.W2_16, HIDDEN * HIDDEN);
  k_cvt_bf16<<<cdiv(HIDDEN * OUTF, 256), 256, 0, stream>>>(Wlin, w.Wl16, HIDDEN * OUTF);

  // Layer 1: gemm(x16,W1)->bufA, att->bufB, bn/elu->bufA (+h16)
  {
    dim3 grid(HIDDEN / 128, N_NODES / 16);
    k_gemm_bf16<F_IN><<<grid, 256, 0, stream>>>(w.x16, w.W1_16, w.bufA, nullptr, HIDDEN);
  }
  run_edge_bn(w.bufA, as1, ad1, b1, bn1g, bn1b, w.bufB, w.bufA, w.h16, ei, w, stream);

  // Layer 2: gemm(h16,W2)->bufB, att->bufA, bn/elu->bufB (+h16, unused after)
  {
    dim3 grid(HIDDEN / 128, N_NODES / 16);
    k_gemm_bf16<HIDDEN><<<grid, 256, 0, stream>>>(w.h16, w.W2_16, w.bufB, nullptr, HIDDEN);
  }
  run_edge_bn(w.bufB, as2, ad2, b2, bn2g, bn2b, w.bufA, w.bufB, w.h16, ei, w, stream);

  // Global mean pool over bufB
  k_pool_zero<<<cdiv(N_GRAPHS * HIDDEN, 256), 256, 0, stream>>>(w.pooled, w.cnt);
  k_pool_acc<<<cdiv(N_NODES * HIDDEN, 256), 256, 0, stream>>>(w.bufB, batch, w.pooled, w.cnt);
  k_pool_div<<<cdiv(N_GRAPHS * HIDDEN, 256), 256, 0, stream>>>(w.pooled, w.cnt, w.p16);

  // Final linear: [1024,128] x [128,256] + blin -> d_out
  {
    dim3 grid(OUTF / 128, N_GRAPHS / 16);
    k_gemm_bf16<HIDDEN><<<grid, 256, 0, stream>>>(w.p16, w.Wl16, out, blin, OUTF);
  }
}
